// BERTxGCN_45543833207355
// MI455X (gfx1250) — matrix operand order repaired and checked by reading, not compiled, not run on hardware
//
#include <hip/hip_runtime.h>
#include <hip/hip_bf16.h>
#include <math.h>

#define HID 768
#define GH  128
#define NC  5
#define NK  (HID / 32)     // 24 K-steps

typedef __attribute__((ext_vector_type(16))) __bf16 v16bf;
typedef __attribute__((ext_vector_type(8)))  float  v8f;

union FragBF { v16bf v; unsigned int u[8]; uint4 q[2]; };

__device__ __forceinline__ unsigned int bf16r(float f) {
  unsigned int u = __float_as_uint(f);
  return (u + 0x7FFFu + ((u >> 16) & 1u)) >> 16;   // round-to-nearest-even
}
__device__ __forceinline__ unsigned int packbf(float a, float b) {
  return (bf16r(a) & 0xFFFFu) | (bf16r(b) << 16);
}

// ==================================================================
// GEMM1: H(bf16)[N,768] = tanh(emb @ Wd + bd)
// 256 thr = 8 waves; block tile 128x128; wave tile 32x64; K-step 32.
// Double-buffered LDS, b128 fragment loads, all B-frags preloaded.
// ==================================================================
__global__ __launch_bounds__(256) void gemm1_kernel(
    const float* __restrict__ A, const float* __restrict__ W,
    const float* __restrict__ bias, unsigned short* __restrict__ H, int N)
{
  __shared__ __align__(16) unsigned int As[2][128][20];  // bf16 pairs As[b][row][k/2]
  __shared__ __align__(16) unsigned int Bs[2][128][20];  // transposed Bs[b][n][k/2]
  const int tid  = threadIdx.x;
  const int lane = tid & 31;
  const int wid  = tid >> 5;
  const int wy   = wid & 3, wx = wid >> 2;     // 4 row-waves x 2 col-waves
  const int rowBlock = blockIdx.y * 128;
  const int colBlock = blockIdx.x * 128;

  v8f acc[2][4] = {};
  float2 aR[8];
  float  bR0[8], bR1[8];

  // ---- prefetch K-tile 0 into registers ----
  #pragma unroll
  for (int i = 0; i < 8; ++i) {
    int idx = tid + i * 256;
    int r = idx >> 4, cu = idx & 15;
    int grow = rowBlock + r; if (grow >= N) grow = N - 1;
    aR[i] = *(const float2*)(A + (size_t)grow * HID + cu * 2);
    int cb = idx >> 7, n = idx & 127;
    bR0[i] = W[(size_t)(2 * cb)     * HID + colBlock + n];
    bR1[i] = W[(size_t)(2 * cb + 1) * HID + colBlock + n];
  }
  // ---- store into LDS buffer 0 ----
  #pragma unroll
  for (int i = 0; i < 8; ++i) {
    int idx = tid + i * 256;
    As[0][idx >> 4][idx & 15] = packbf(aR[i].x, aR[i].y);
    Bs[0][idx & 127][idx >> 7] = packbf(bR0[i], bR1[i]);
  }

  int cur = 0;
  const int arow = wy * 32 + (lane & 15);
  const int ka4  = (lane < 16) ? 0 : 1;        // uint4 index (K 0..7 vs 8..15)
  const int kb4  = (lane < 16) ? 0 : 1;        // uint4 index (K 0..7 vs 8..15)

  for (int kt = 0; kt < NK; ++kt) {
    __syncthreads();

    // prefetch next K-tile (global -> regs) while computing
    if (kt + 1 < NK) {
      const int k0 = (kt + 1) * 32;
      #pragma unroll
      for (int i = 0; i < 8; ++i) {
        int idx = tid + i * 256;
        int r = idx >> 4, cu = idx & 15;
        int grow = rowBlock + r; if (grow >= N) grow = N - 1;
        aR[i] = *(const float2*)(A + (size_t)grow * HID + k0 + cu * 2);
        int cb = idx >> 7, n = idx & 127;
        bR0[i] = W[(size_t)(k0 + 2 * cb)     * HID + colBlock + n];
        bR1[i] = W[(size_t)(k0 + 2 * cb + 1) * HID + colBlock + n];
      }
    }

    // ---- load ALL fragments first (lets DS waits pipeline), then 8 WMMAs ----
    FragBF a0, a1, b[4];
    {
      const uint4* rv0 = (const uint4*)&As[cur][arow][0];
      const uint4* rv1 = (const uint4*)&As[cur][arow + 16][0];
      a0.q[0] = rv0[ka4]; a0.q[1] = rv0[ka4 + 2];
      a1.q[0] = rv1[ka4]; a1.q[1] = rv1[ka4 + 2];
    }
    #pragma unroll
    for (int t = 0; t < 4; ++t) {
      const uint4* bv = (const uint4*)&Bs[cur][wx * 64 + t * 16 + (lane & 15)][0];
      b[t].q[0] = bv[kb4]; b[t].q[1] = bv[kb4 + 2];
    }
    #pragma unroll
    for (int t = 0; t < 4; ++t) {
      acc[0][t] = __builtin_amdgcn_wmma_f32_16x16x32_bf16(
          false, a0.v, false, b[t].v, (short)0, acc[0][t], false, false);
      acc[1][t] = __builtin_amdgcn_wmma_f32_16x16x32_bf16(
          false, a1.v, false, b[t].v, (short)0, acc[1][t], false, false);
    }

    // store prefetched tile into the other buffer
    if (kt + 1 < NK) {
      #pragma unroll
      for (int i = 0; i < 8; ++i) {
        int idx = tid + i * 256;
        As[cur ^ 1][idx >> 4][idx & 15] = packbf(aR[i].x, aR[i].y);
        Bs[cur ^ 1][idx & 127][idx >> 7] = packbf(bR0[i], bR1[i]);
      }
    }
    cur ^= 1;
  }

  // ---- epilogue: bias + tanh -> bf16 ----
  #pragma unroll
  for (int t = 0; t < 4; ++t) {
    const int col = colBlock + wx * 64 + t * 16 + (lane & 15);
    const float bv = bias[col];
    #pragma unroll
    for (int m = 0; m < 2; ++m) {
      #pragma unroll
      for (int r = 0; r < 8; ++r) {
        const int row = rowBlock + wy * 32 + m * 16 + ((lane < 16) ? r : r + 8);
        if (row < N) {
          float v = tanhf(acc[m][t][r] + bv);
          H[(size_t)row * HID + col] = (unsigned short)(bf16r(v) & 0xFFFFu);
        }
      }
    }
  }
}

// ==================================================================
// GEMM2: xw[N,128] = [h|p_num|text_len] @ W1  (rank-1 epilogue)
// Same pipeline; A is already bf16.
// ==================================================================
__global__ __launch_bounds__(256) void gemm2_kernel(
    const unsigned short* __restrict__ Hb, const float* __restrict__ W1,
    const float* __restrict__ p_num, const float* __restrict__ text_len,
    float* __restrict__ xw, int N)
{
  __shared__ __align__(16) unsigned int As[2][128][20];
  __shared__ __align__(16) unsigned int Bs[2][128][20];
  const int tid  = threadIdx.x;
  const int lane = tid & 31;
  const int wid  = tid >> 5;
  const int wy   = wid & 3, wx = wid >> 2;
  const int rowBlock = blockIdx.y * 128;

  v8f acc[2][4] = {};
  unsigned int aR[8];
  float bR0[8], bR1[8];

  #pragma unroll
  for (int i = 0; i < 8; ++i) {
    int idx = tid + i * 256;
    int r = idx >> 4, cu = idx & 15;
    int grow = rowBlock + r; if (grow >= N) grow = N - 1;
    aR[i] = *(const unsigned int*)(Hb + (size_t)grow * HID + cu * 2);
    int cb = idx >> 7, n = idx & 127;
    bR0[i] = W1[(size_t)(2 * cb)     * GH + n];
    bR1[i] = W1[(size_t)(2 * cb + 1) * GH + n];
  }
  #pragma unroll
  for (int i = 0; i < 8; ++i) {
    int idx = tid + i * 256;
    As[0][idx >> 4][idx & 15] = aR[i];
    Bs[0][idx & 127][idx >> 7] = packbf(bR0[i], bR1[i]);
  }

  int cur = 0;
  const int arow = wy * 32 + (lane & 15);
  const int ka4  = (lane < 16) ? 0 : 1;
  const int kb4  = (lane < 16) ? 0 : 1;

  for (int kt = 0; kt < NK; ++kt) {
    __syncthreads();

    if (kt + 1 < NK) {
      const int k0 = (kt + 1) * 32;
      #pragma unroll
      for (int i = 0; i < 8; ++i) {
        int idx = tid + i * 256;
        int r = idx >> 4, cu = idx & 15;
        int grow = rowBlock + r; if (grow >= N) grow = N - 1;
        aR[i] = *(const unsigned int*)(Hb + (size_t)grow * HID + k0 + cu * 2);
        int cb = idx >> 7, n = idx & 127;
        bR0[i] = W1[(size_t)(k0 + 2 * cb)     * GH + n];
        bR1[i] = W1[(size_t)(k0 + 2 * cb + 1) * GH + n];
      }
    }

    FragBF a0, a1, b[4];
    {
      const uint4* rv0 = (const uint4*)&As[cur][arow][0];
      const uint4* rv1 = (const uint4*)&As[cur][arow + 16][0];
      a0.q[0] = rv0[ka4]; a0.q[1] = rv0[ka4 + 2];
      a1.q[0] = rv1[ka4]; a1.q[1] = rv1[ka4 + 2];
    }
    #pragma unroll
    for (int t = 0; t < 4; ++t) {
      const uint4* bv = (const uint4*)&Bs[cur][wx * 64 + t * 16 + (lane & 15)][0];
      b[t].q[0] = bv[kb4]; b[t].q[1] = bv[kb4 + 2];
    }
    #pragma unroll
    for (int t = 0; t < 4; ++t) {
      acc[0][t] = __builtin_amdgcn_wmma_f32_16x16x32_bf16(
          false, a0.v, false, b[t].v, (short)0, acc[0][t], false, false);
      acc[1][t] = __builtin_amdgcn_wmma_f32_16x16x32_bf16(
          false, a1.v, false, b[t].v, (short)0, acc[1][t], false, false);
    }

    if (kt + 1 < NK) {
      #pragma unroll
      for (int i = 0; i < 8; ++i) {
        int idx = tid + i * 256;
        As[cur ^ 1][idx >> 4][idx & 15] = aR[i];
        Bs[cur ^ 1][idx & 127][idx >> 7] = packbf(bR0[i], bR1[i]);
      }
    }
    cur ^= 1;
  }

  // epilogue: add p_num/text_len rank-1 contributions (concat columns 768,769)
  #pragma unroll
  for (int t = 0; t < 4; ++t) {
    const int col = wx * 64 + t * 16 + (lane & 15);
    const float w1a = W1[(size_t)HID * GH + col];
    const float w1b = W1[(size_t)(HID + 1) * GH + col];
    #pragma unroll
    for (int m = 0; m < 2; ++m) {
      #pragma unroll
      for (int r = 0; r < 8; ++r) {
        const int row = rowBlock + wy * 32 + m * 16 + ((lane < 16) ? r : r + 8);
        if (row < N) {
          xw[(size_t)row * GH + col] =
              acc[m][t][r] + p_num[row] * w1a + text_len[row] * w1b;
        }
      }
    }
  }
}

// ============ GCN normalization helpers ============
__global__ void deg_init(float* deg, int N) {
  int i = blockIdx.x * blockDim.x + threadIdx.x;
  if (i < N) deg[i] = 2.0f;                       // improved=True self-loop weight
}
__global__ void deg_scatter(const int* __restrict__ col, const float* __restrict__ w,
                            float* deg, int E) {
  int e = blockIdx.x * blockDim.x + threadIdx.x;
  if (e < E) atomicAdd(&deg[col[e]], w[e]);
}
__global__ void dis_kernel(const float* __restrict__ deg, float* dis, int N) {
  int i = blockIdx.x * blockDim.x + threadIdx.x;
  if (i < N) { float d = deg[i]; dis[i] = (d > 0.0f) ? rsqrtf(d) : 0.0f; }
}
__global__ void norm_kernel(const int* __restrict__ row, const int* __restrict__ col,
                            const float* __restrict__ w, const float* __restrict__ dis,
                            float* nrm, int E) {
  int e = blockIdx.x * blockDim.x + threadIdx.x;
  if (e < E) nrm[e] = dis[row[e]] * w[e] * dis[col[e]];
}

// ============ Layer-1 aggregation ============
__global__ void agg1_init(const float* __restrict__ xw, const float* __restrict__ dis,
                          const float* __restrict__ b1, float* agg, int N) {
  int idx = blockIdx.x * blockDim.x + threadIdx.x;
  if (idx < N * GH) {
    int i = idx >> 7, c = idx & 127;
    float s = dis[i];
    agg[idx] = b1[c] + 2.0f * s * s * xw[idx];    // self-loop term + bias
  }
}
__global__ __launch_bounds__(256) void agg1_edges(
    const int* __restrict__ row, const int* __restrict__ col,
    const float* __restrict__ nrm, const float* __restrict__ xw,
    float* agg, int E) {
  int e = blockIdx.x * 2 + (threadIdx.x >> 7);    // 2 edges per block
  int c = threadIdx.x & 127;
  if (e < E) {
    float nv = nrm[e];
    atomicAdd(&agg[(size_t)col[e] * GH + c], nv * xw[(size_t)row[e] * GH + c]);
  }
}

// ============ relu + tiny layer-2 transform ============
__global__ void y2_kernel(const float* __restrict__ agg, const float* __restrict__ W2,
                          float* y2, int N) {
  int idx = blockIdx.x * blockDim.x + threadIdx.x;
  if (idx >= N * NC) return;
  int i = idx / NC, j = idx - i * NC;
  const float* a = agg + (size_t)i * GH;
  float s = 0.0f;
  #pragma unroll 4
  for (int c = 0; c < GH; ++c) {
    float x = a[c]; x = (x > 0.0f) ? x : 0.0f;
    s += x * W2[c * NC + j];
  }
  y2[idx] = s;
}

// ============ Layer-2 aggregation into d_out ============
__global__ void out_init(const float* __restrict__ y2, const float* __restrict__ dis,
                         const float* __restrict__ b2, float* out, int N) {
  int idx = blockIdx.x * blockDim.x + threadIdx.x;
  if (idx < N * NC) {
    int i = idx / NC, j = idx - i * NC;
    float s = dis[i];
    out[idx] = b2[j] + 2.0f * s * s * y2[idx];
  }
}
__global__ void out_edges(const int* __restrict__ row, const int* __restrict__ col,
                          const float* __restrict__ nrm, const float* __restrict__ y2,
                          float* out, int E) {
  long idx = (long)blockIdx.x * blockDim.x + threadIdx.x;
  if (idx >= (long)E * NC) return;
  int e = (int)(idx / NC), j = (int)(idx - (long)e * NC);
  atomicAdd(&out[(size_t)col[e] * NC + j], nrm[e] * y2[(size_t)row[e] * NC + j]);
}

extern "C" void kernel_launch(void* const* d_in, const int* in_sizes, int n_in,
                              void* d_out, int out_size, void* d_ws, size_t ws_size,
                              hipStream_t stream) {
  const float* emb      = (const float*)d_in[0];
  const float* p_num    = (const float*)d_in[1];
  const float* text_len = (const float*)d_in[2];
  const int*   eidx     = (const int*)d_in[3];
  const float* ew       = (const float*)d_in[4];
  const float* Wd       = (const float*)d_in[5];
  const float* bd       = (const float*)d_in[6];
  const float* W1       = (const float*)d_in[7];
  const float* b1       = (const float*)d_in[8];
  const float* W2       = (const float*)d_in[9];
  const float* b2       = (const float*)d_in[10];
  const int N = in_sizes[1];                 // p_num has N elements
  const int E = in_sizes[4];                 // edge_weight has E elements
  const int* erow = eidx;
  const int* ecol = eidx + E;

  char* ws = (char*)d_ws;
  size_t off = 0;
  auto alloc = [&](size_t bytes) -> void* {
    void* p = ws + off; off += (bytes + 255) & ~(size_t)255; return p;
  };
  unsigned short* hbuf = (unsigned short*)alloc((size_t)N * HID * 2); // bf16 h
  float* xw  = (float*)alloc((size_t)N * GH * 4);
  float* agg = (float*)alloc((size_t)N * GH * 4);
  float* y2  = (float*)alloc((size_t)N * NC * 4);
  float* deg = (float*)alloc((size_t)N * 4);
  float* dis = (float*)alloc((size_t)N * 4);
  float* nrm = (float*)alloc((size_t)E * 4);

  // GCN normalization
  deg_init   <<<(N + 255) / 256, 256, 0, stream>>>(deg, N);
  deg_scatter<<<(E + 255) / 256, 256, 0, stream>>>(ecol, ew, deg, E);
  dis_kernel <<<(N + 255) / 256, 256, 0, stream>>>(deg, dis, N);
  norm_kernel<<<(E + 255) / 256, 256, 0, stream>>>(erow, ecol, ew, dis, nrm, E);

  // dense + tanh (WMMA bf16), then x@W1 with concat folded as rank-1 epilogue
  gemm1_kernel<<<dim3(HID / 128, (N + 127) / 128), 256, 0, stream>>>(emb, Wd, bd, hbuf, N);
  gemm2_kernel<<<dim3(1, (N + 127) / 128), 256, 0, stream>>>(hbuf, W1, p_num, text_len, xw, N);

  // layer 1 aggregation, relu fused into tiny layer-2 transform
  agg1_init <<<(int)(((size_t)N * GH + 255) / 256), 256, 0, stream>>>(xw, dis, b1, agg, N);
  agg1_edges<<<(E + 1) / 2, 256, 0, stream>>>(erow, ecol, nrm, xw, agg, E);
  y2_kernel <<<(int)(((size_t)N * NC + 255) / 256), 256, 0, stream>>>(agg, W2, y2, N);

  // layer 2 aggregation directly into d_out (fully initialized every call)
  out_init <<<(int)(((size_t)N * NC + 255) / 256), 256, 0, stream>>>(y2, dis, b2, (float*)d_out, N);
  out_edges<<<(int)(((size_t)E * NC + 255) / 256), 256, 0, stream>>>(erow, ecol, nrm, y2, (float*)d_out, E);
}
